// Seq2Seq_41747082117786
// MI455X (gfx1250) — matrix operand order repaired and checked by reading, not compiled, not run on hardware
//
#include <hip/hip_runtime.h>
#include <cmath>

typedef __attribute__((ext_vector_type(16))) _Float16 v16h;
typedef __attribute__((ext_vector_type(8)))  _Float16 v8h;
typedef __attribute__((ext_vector_type(4)))  _Float16 v4h;
typedef __attribute__((ext_vector_type(8)))  float    v8f;

#define DEVI __device__ __forceinline__
#define CAT16(lo,hi) __builtin_shufflevector(lo,hi,0,1,2,3,4,5,6,7,8,9,10,11,12,13,14,15)

namespace {
constexpr int kB = 8192, kT = 100, kIN = 8, kH = 128, kG = 512, kAK = 160, kOUT = 30;
constexpr int kThreads = 256, kWaves = 8, kRowsPerWave = 16;
constexpr int kBlockRows = kWaves * kRowsPerWave;           // 128 batch rows / block
constexpr int kBlocks = kB / kBlockRows;                    // 64 blocks

// LDS layout (bytes). Bias is folded into column k==8 of the weight tiles
// (A-tile column 8 is a constant 1.0), so no separate bias arrays.
constexpr size_t offWc   = 0;                               // [G][AK] f16: [Wih | bias | 0 | Whh]
constexpr size_t offFcw  = offWc  + (size_t)kG * kAK * 2;   // [16][AK] f16: [0 | fc_b | 0 | fc_W] (rows 8..15 zero)
constexpr size_t offA    = offFcw + (size_t)16 * kAK * 2;   // [waves][16][AK] f16 A-tiles ([x|1|0|h])
constexpr size_t kSmemBytes = offA + (size_t)kWaves * 16 * kAK * 2;  // ~205 KB (fits 320KB WGP LDS)
} // namespace

DEVI float fast_tanh(float x) {
#if __has_builtin(__builtin_amdgcn_tanhf)
  return __builtin_amdgcn_tanhf(x);          // v_tanh_f32 (CDNA5 native TRANS)
#else
  float r;
  asm("v_tanh_f32 %0, %1" : "=v"(r) : "v"(x));
  return r;
#endif
}
DEVI float sigf(float x) {                   // sigmoid(x) = 0.5*tanh(x/2) + 0.5
  return fmaf(0.5f, fast_tanh(0.5f * x), 0.5f);
}

// Build A fragments (16x32 f16 per k-step) per ISA layout:
// lane<16: elements 0..7 = K base+0..7, 8..15 = K base+16..23
// lane>=16: elements 0..7 = K base+8..15, 8..15 = K base+24..31
template <int NK>
DEVI void load_afrag(const _Float16* arow, int hiSel, int kOff, v16h* a) {
#pragma unroll
  for (int kk = 0; kk < NK; ++kk) {
    const _Float16* p = arow + kOff + 32 * kk + hiSel * 8;
    v8h lo = *(const v8h*)p;
    v8h hi = *(const v8h*)(p + 16);
    a[kk] = CAT16(lo, hi);
  }
}

// B fragment (32x16 f16): lane covers column N = lane&15, same K order as A
// (lanes<16: K base+0..7,16..23; lanes>=16: K base+8..15,24..31). wcol points at the
// column's contiguous K row in LDS. acc starts as zeroinitializer -> inline SRC2=0.
template <int NK>
DEVI v8f gemm_tile(const _Float16* wcol, int hiSel, const v16h* a) {
  v8f acc = {0.f, 0.f, 0.f, 0.f, 0.f, 0.f, 0.f, 0.f};
#pragma unroll
  for (int kk = 0; kk < NK; ++kk) {
    const _Float16* p = wcol + 32 * kk + hiSel * 8;
    v8h lo = *(const v8h*)p;
    v8h hi = *(const v8h*)(p + 16);
    v16h b = CAT16(lo, hi);
    acc = __builtin_amdgcn_wmma_f32_16x16x32_f16(false, a[kk], false, b,
                                                 (short)0, acc, false, false);
  }
  return acc;
}

// one LSTM cell step for a 16-row tile: gates via WMMA (bias folded into k==8
// column of Wc), state update with native v_tanh_f32, h -> LDS (f16)
DEVI void lstm_step(const _Float16* Wc, const v16h* afrag,
                    float (&cst)[8][8], _Float16* myA, int hiSel, int colL) {
#pragma unroll
  for (int j = 0; j < 8; ++j) {
    v8f gi = gemm_tile<5>(Wc + (size_t)(16 * j + colL) * kAK, hiSel, afrag);
    v8f gf = gemm_tile<5>(Wc + (size_t)(16 * (8 + j) + colL) * kAK, hiSel, afrag);
    v8f gg = gemm_tile<5>(Wc + (size_t)(16 * (16 + j) + colL) * kAK, hiSel, afrag);
    v8f go = gemm_tile<5>(Wc + (size_t)(16 * (24 + j) + colL) * kAK, hiSel, afrag);
#pragma unroll
    for (int r = 0; r < 8; ++r) {
      float iv = sigf(gi[r]);
      float fv = sigf(gf[r]);
      float gv = fast_tanh(gg[r]);
      float ov = sigf(go[r]);
      float cn = fv * cst[j][r] + iv * gv;
      cst[j][r] = cn;
      // D layout: VGPR r -> row r + 8*hiSel, col = lane&15; h lives at K = 32 + col16
      myA[(r + 8 * hiSel) * kAK + 32 + 16 * j + colL] = (_Float16)(ov * fast_tanh(cn));
    }
  }
}

__global__ __launch_bounds__(kThreads, 1)
void seq2seq_lstm_wmma(const float* __restrict__ x,
                       const float* __restrict__ eWih, const float* __restrict__ eWhh,
                       const float* __restrict__ eBih, const float* __restrict__ eBhh,
                       const float* __restrict__ dWih, const float* __restrict__ dWhh,
                       const float* __restrict__ dBih, const float* __restrict__ dBhh,
                       const float* __restrict__ fcW, const float* __restrict__ fcB,
                       float* __restrict__ out) {
  extern __shared__ __align__(16) char smem[];
  _Float16* Wc  = (_Float16*)(smem + offWc);
  _Float16* fcw = (_Float16*)(smem + offFcw);
  _Float16* Atl = (_Float16*)(smem + offA);

  const int tid   = threadIdx.x;
  const int lane  = tid & 31;
  const int wave  = tid >> 5;
  const int hiSel = lane >> 4;   // 0: lanes 0-15, 1: lanes 16-31
  const int colL  = lane & 15;
  const int bRow0 = blockIdx.x * kBlockRows + wave * kRowsPerWave;

  _Float16* myA = Atl + (size_t)wave * 16 * kAK;     // this wave's [x|1|0|h] tile
  const _Float16* arow = myA + (size_t)colL * kAK;   // A-operand row for this lane

  // ---- phase 0: encoder weights -> LDS: Wc[n] = [Wih(8) | bias | 0(23) | Whh(128)] ----
  for (int idx = tid; idx < kG * kAK; idx += kThreads) {
    int n = idx / kAK, k = idx - n * kAK;
    float v = (k < kIN) ? eWih[n * kIN + k]
            : (k == kIN) ? (eBih[n] + eBhh[n])
            : (k >= 32) ? eWhh[n * kH + (k - 32)]
            : 0.f;
    Wc[idx] = (_Float16)v;
  }
  // fc tile: fcw[n] = [0(8) | fc_b | 0(23) | fc_W(128)], rows 8..15 all zero
  for (int idx = tid; idx < 16 * kAK; idx += kThreads) {
    int n = idx / kAK, k = idx - n * kAK;
    float v = 0.f;
    if (n < kIN) {
      if (k == kIN) v = fcB[n];
      else if (k >= 32) v = fcW[n * kH + (k - 32)];
    }
    fcw[idx] = (_Float16)v;
  }
  // A-tiles: zero everywhere except constant 1.0 at k==8 (bias lane)
  for (int idx = tid; idx < kWaves * 16 * kAK; idx += kThreads) {
    int k = idx % kAK;
    Atl[idx] = (_Float16)((k == kIN) ? 1.f : 0.f);
  }
  __syncthreads();

  float cst[8][8];  // c-state: [j-tile][row-vgpr] for this lane's (row,col) coverage
#pragma unroll
  for (int j = 0; j < 8; ++j)
#pragma unroll
    for (int r = 0; r < 8; ++r) cst[j][r] = 0.f;

  // per-lane x_t loader: lane covers row = lane>>1, 4 contiguous features
  const int   xrow  = lane >> 1;
  const int   xhalf = lane & 1;
  const float* xbase = x + ((size_t)(bRow0 + xrow) * kT) * kIN + xhalf * 4;

  // ---------------- encoder: 100 recurrent steps ----------------
  for (int t = 0; t < kT; ++t) {
    const float4 xv = *(const float4*)(xbase + (size_t)t * kIN);
    if (t + 1 < kT) __builtin_prefetch(xbase + (size_t)(t + 1) * kIN, 0, 1);
    v4h xh = {(_Float16)xv.x, (_Float16)xv.y, (_Float16)xv.z, (_Float16)xv.w};
    *(v4h*)(myA + (size_t)xrow * kAK + xhalf * 4) = xh;   // x_t into K[0..8)
    __syncthreads();  // x_t store + prev h store visible to A-fragment loads

    v16h afrag[5];
    load_afrag<5>(arow, hiSel, 0, afrag);
    lstm_step(Wc, afrag, cst, myA, hiSel, colL);
  }

  // ---- phase switch: decoder weights into the SAME 160KB LDS region ----
  __syncthreads();
  for (int idx = tid; idx < kG * kAK; idx += kThreads) {
    int n = idx / kAK, k = idx - n * kAK;
    float v = (k < kIN) ? dWih[n * kIN + k]
            : (k == kIN) ? (dBih[n] + dBhh[n])
            : (k >= 32) ? dWhh[n * kH + (k - 32)]
            : 0.f;
    Wc[idx] = (_Float16)v;
  }
  // NOTE: A-tile untouched: K[0..8) still holds x[:,T-1,:] == dec_in,
  // k==8 still 1.0, K[32..) holds h_enc

  // ---------------- decoder: 30 autoregressive steps ----------------
  for (int t = 0; t < kOUT; ++t) {
    __syncthreads();  // prev y/h stores (and weight reload at t==0) visible

    v16h afrag[5];
    load_afrag<5>(arow, hiSel, 0, afrag);
    lstm_step(Wc, afrag, cst, myA, hiSel, colL);

    __syncthreads();  // h_new visible for fc GEMM
    v16h hfrag[5];
    load_afrag<5>(arow, hiSel, 0, hfrag);  // [y_old|1|0|h_new]; fcw zeros kill K<8
    v8f y = gemm_tile<5>(fcw + (size_t)colL * kAK, hiSel, hfrag);  // bias via k==8

    if (colL < kIN) {  // cols 8..15 are zero padding
#pragma unroll
      for (int r = 0; r < 8; ++r) {
        const int row = r + 8 * hiSel;
        out[((size_t)(bRow0 + row) * kOUT + t) * kIN + colL] = y[r];
        myA[(size_t)row * kAK + colL] = (_Float16)y[r];  // feed y back as next input
      }
    }
  }
}

extern "C" void kernel_launch(void* const* d_in, const int* in_sizes, int n_in,
                              void* d_out, int out_size, void* d_ws, size_t ws_size,
                              hipStream_t stream) {
  (void)in_sizes; (void)n_in; (void)out_size; (void)d_ws; (void)ws_size;
  const float* x    = (const float*)d_in[0];
  const float* eWih = (const float*)d_in[1];
  const float* eWhh = (const float*)d_in[2];
  const float* eBih = (const float*)d_in[3];
  const float* eBhh = (const float*)d_in[4];
  const float* dWih = (const float*)d_in[5];
  const float* dWhh = (const float*)d_in[6];
  const float* dBih = (const float*)d_in[7];
  const float* dBhh = (const float*)d_in[8];
  const float* fcW  = (const float*)d_in[9];
  const float* fcB  = (const float*)d_in[10];
  float* out = (float*)d_out;

  // ~205 KB dynamic LDS (CDNA5 WGP has 320 KB) — raise the per-kernel cap.
  (void)hipFuncSetAttribute((const void*)seq2seq_lstm_wmma,
                            hipFuncAttributeMaxDynamicSharedMemorySize,
                            (int)kSmemBytes);

  seq2seq_lstm_wmma<<<dim3(kBlocks), dim3(kThreads), kSmemBytes, stream>>>(
      x, eWih, eWhh, eBih, eBhh, dWih, dWhh, dBih, dBhh, fcW, fcB, out);
}
